// GatedLinearRecurrence_36490042146961
// MI455X (gfx1250) — matrix operand order repaired and checked by reading, not compiled, not run on hardware
//
#include <hip/hip_runtime.h>

// ---------- types ----------
typedef __attribute__((ext_vector_type(16))) __bf16          v16bf;
typedef __attribute__((ext_vector_type(8)))  float           v8f;
typedef __attribute__((ext_vector_type(4)))  unsigned int    v4u;
typedef __attribute__((ext_vector_type(4)))  int             v4i;
typedef __attribute__((ext_vector_type(4)))  unsigned short  v4us;
typedef __attribute__((ext_vector_type(4)))  float           v4f;

union FragBF { v16bf bf; v4u q[2]; };

// ---------- CDNA5 async global->LDS copy (ASYNCcnt path), with sync fallback ----------
#if defined(__AMDGCN__) && __has_builtin(__builtin_amdgcn_global_load_async_to_lds_b128)
#define HAVE_ASYNC 1
#else
#define HAVE_ASYNC 0
#if defined(__AMDGCN__)
#warning "gfx1250 async-to-LDS builtin not available; using synchronous LDS staging"
#endif
#endif

__device__ __forceinline__ void async_cp16(const unsigned short* g, unsigned short* l) {
#if HAVE_ASYNC
  __builtin_amdgcn_global_load_async_to_lds_b128(
      (__attribute__((address_space(1))) v4i*)(v4i*)g,
      (__attribute__((address_space(3))) v4i*)(v4i*)l,
      0, 0);
#else
  *(v4u*)l = *(const v4u*)g;
#endif
}

__device__ __forceinline__ void wait_async0() {
#if HAVE_ASYNC
#if __has_builtin(__builtin_amdgcn_s_wait_asynccnt)
  __builtin_amdgcn_s_wait_asynccnt(0);
#else
  asm volatile("s_wait_asynccnt 0x0" ::: "memory");
#endif
#endif
}

__device__ __forceinline__ unsigned short f2bf(float f) {
  unsigned int u = __float_as_uint(f);
  u += 0x7fffu + ((u >> 16) & 1u);   // round-to-nearest-even
  return (unsigned short)(u >> 16);
}

__device__ __forceinline__ float sigmoidf_(float v) { return 1.0f / (1.0f + __expf(-v)); }
__device__ __forceinline__ float siluf_(float v)    { return v / (1.0f + __expf(-v)); }

// ---------- constants ----------
#define DMODEL 1024
#define DINNER 2048
#define LSEQ   4096
#define BATCH  2
#define MTOT   (BATCH * LSEQ)   // 8192 rows
#define NCHUNK 16
#define CLEN   (LSEQ / NCHUNK)  // 256

// ---------- fp32 -> bf16 convert (weights) ----------
__global__ __launch_bounds__(256) void cvt_bf16(const float* __restrict__ src,
                                                unsigned short* __restrict__ dst, int n4) {
  int i = blockIdx.x * 256 + threadIdx.x;
  if (i >= n4) return;
  v4f v = *(const v4f*)(src + (size_t)i * 4);
  v4us o;
  #pragma unroll
  for (int j = 0; j < 4; ++j) o[j] = f2bf(v[j]);
  *(v4us*)(dst + (size_t)i * 4) = o;
}

// ---------- LayerNorm (one wave per row of 1024) -> bf16 ----------
__global__ __launch_bounds__(256) void ln_bf16(const float* __restrict__ x,
                                               const float* __restrict__ gw,
                                               const float* __restrict__ gb,
                                               unsigned short* __restrict__ out, int rows) {
  int row  = blockIdx.x * 8 + (threadIdx.x >> 5);
  int lane = threadIdx.x & 31;
  if (row >= rows) return;
  const float* xr = x + (size_t)row * DMODEL;
  float s = 0.f, s2 = 0.f;
  v4f v[8];
  #pragma unroll
  for (int i = 0; i < 8; ++i) {
    v[i] = *(const v4f*)(xr + (i * 32 + lane) * 4);
    #pragma unroll
    for (int j = 0; j < 4; ++j) { float f = v[i][j]; s += f; s2 += f * f; }
  }
  #pragma unroll
  for (int off = 16; off > 0; off >>= 1) {
    s  += __shfl_xor(s,  off, 32);
    s2 += __shfl_xor(s2, off, 32);
  }
  float mean = s * (1.0f / DMODEL);
  float var  = s2 * (1.0f / DMODEL) - mean * mean;
  float rstd = rsqrtf(var + 1e-5f);
  unsigned short* orow = out + (size_t)row * DMODEL;
  #pragma unroll
  for (int i = 0; i < 8; ++i) {
    int c0 = (i * 32 + lane) * 4;
    v4f wv = *(const v4f*)(gw + c0);
    v4f bv = *(const v4f*)(gb + c0);
    v4us o;
    #pragma unroll
    for (int j = 0; j < 4; ++j) o[j] = f2bf((v[i][j] - mean) * rstd * wv[j] + bv[j]);
    *(v4us*)(orow + c0) = o;
  }
}

// ---------- WMMA bf16 GEMM: C[M,N] = A[M,K] * W[N,K]^T, fused epilogues ----------
// MODE 0: split store (col<N/2 -> out0, else out1), both [M, N/2]
// MODE 1: a = sigmoid(acc + aux0[col]); out0 = a; out1 = (1-a) * aux1[row*N+col]
// MODE 2: out0[row*N+col] = aux0[row*N+col] + acc  (residual add)
#define TM 128
#define TN 128
#define TK 32
#define LDST 40   // padded LDS row: 32 bf16 data + 8 pad (80 B -> conflict-free b128 frags)

template <int MODE>
__global__ __launch_bounds__(256) void gemm_bf16(
    const unsigned short* __restrict__ A, const unsigned short* __restrict__ W,
    int M, int N, int K,
    float* __restrict__ out0, float* __restrict__ out1,
    const float* __restrict__ aux0, const float* __restrict__ aux1) {
  __shared__ unsigned short la[2][TM * LDST];
  __shared__ unsigned short lb[2][TN * LDST];

  const int tid   = threadIdx.x;
  const int lane  = tid & 31;
  const int wv    = tid >> 5;
  const int waveM = wv >> 1;  // 0..3 -> 32-row strips
  const int waveN = wv & 1;   // 0..1 -> 64-col strips
  const int rowBase = blockIdx.y * TM;
  const int colBase = blockIdx.x * TN;
  const int half    = lane >> 4;
  const int l15     = lane & 15;

  // per-thread staging chunks (2 x b128 per tile per thread)
  const int sr = tid >> 2, sq = tid & 3;
  const int sr2 = (tid + 256) >> 2, sq2 = (tid + 256) & 3;

  auto stage = [&](int kk, int buf) {
    async_cp16(A + (size_t)(rowBase + sr)  * K + kk + sq  * 8, &la[buf][sr  * LDST + sq  * 8]);
    async_cp16(W + (size_t)(colBase + sr)  * K + kk + sq  * 8, &lb[buf][sr  * LDST + sq  * 8]);
    async_cp16(A + (size_t)(rowBase + sr2) * K + kk + sq2 * 8, &la[buf][sr2 * LDST + sq2 * 8]);
    async_cp16(W + (size_t)(colBase + sr2) * K + kk + sq2 * 8, &lb[buf][sr2 * LDST + sq2 * 8]);
  };

  v8f acc[2][4];
  #pragma unroll
  for (int i = 0; i < 2; ++i)
    #pragma unroll
    for (int j = 0; j < 4; ++j) acc[i][j] = (v8f){0, 0, 0, 0, 0, 0, 0, 0};

  // prologue: stage first K-tile into buffer 0
  stage(0, 0);
  wait_async0();
  __syncthreads();

  int cur = 0;
  for (int k0 = 0; k0 < K; k0 += TK) {
    // start DMA of next tile into the other buffer (overlaps with WMMAs below)
    if (k0 + TK < K) stage(k0 + TK, cur ^ 1);

    FragBF af[2];
    #pragma unroll
    for (int mt = 0; mt < 2; ++mt) {
      int r = waveM * 32 + mt * 16 + l15;
      int base = r * LDST + half * 8;           // lanes>=16 read K 8..15 / 24..31
      af[mt].q[0] = *(const v4u*)&la[cur][base];
      af[mt].q[1] = *(const v4u*)&la[cur][base + 16];
    }
    #pragma unroll
    for (int nt = 0; nt < 4; ++nt) {
      int n = waveN * 64 + nt * 16 + l15;
      int base = n * LDST + half * 8;           // lanes>=16 read K 8..15 / 24..31
      FragBF bf;
      bf.q[0] = *(const v4u*)&lb[cur][base];
      bf.q[1] = *(const v4u*)&lb[cur][base + 16];
      #pragma unroll
      for (int mt = 0; mt < 2; ++mt)
        acc[mt][nt] = __builtin_amdgcn_wmma_f32_16x16x32_bf16(
            false, af[mt].bf, false, bf.bf, (short)0, acc[mt][nt], false, false);
    }

    // my async writes to buf^1 done; barrier -> everyone's writes visible,
    // and everyone is done reading buf (so it can be overwritten next iter)
    wait_async0();
    __syncthreads();
    cur ^= 1;
  }

  // epilogue: VGPR r holds M=r (lanes 0-15) and M=r+8 (lanes 16-31), N=lane%16
  #pragma unroll
  for (int mt = 0; mt < 2; ++mt) {
    #pragma unroll
    for (int nt = 0; nt < 4; ++nt) {
      int col  = colBase + waveN * 64 + nt * 16 + l15;
      int row0 = rowBase + waveM * 32 + mt * 16 + (half << 3);
      #pragma unroll
      for (int r = 0; r < 8; ++r) {
        int row = row0 + r;
        float v = acc[mt][nt][r];
        if (MODE == 0) {
          int nh = N >> 1;
          if (col < nh) out0[(size_t)row * nh + col] = v;
          else          out1[(size_t)row * nh + col - nh] = v;
        } else if (MODE == 1) {
          float g  = sigmoidf_(v + aux0[col]);
          size_t o = (size_t)row * N + col;
          out0[o] = g;
          out1[o] = (1.0f - g) * aux1[o];
        } else {
          size_t o = (size_t)row * N + col;
          out0[o] = aux0[o] + v;
        }
      }
    }
  }
}

// ---------- causal depthwise conv (K=4) + bias + SiLU ----------
__global__ __launch_bounds__(256) void conv_silu(const float* __restrict__ xin,
                                                 const float* __restrict__ cw,
                                                 const float* __restrict__ cb,
                                                 float* __restrict__ xc,
                                                 unsigned short* __restrict__ xcb) {
  int idx = blockIdx.x * 256 + threadIdx.x;  // over B*L*C = 2^24
  int c  = idx & (DINNER - 1);
  int t  = (idx >> 11) & (LSEQ - 1);
  int bi = idx >> 23;
  float v = cb[c];
  #pragma unroll
  for (int k = 0; k < 4; ++k) {
    int tt = t - 3 + k;
    if (tt >= 0)
      v += cw[c * 4 + k] * xin[(((size_t)bi << 12) + tt) * DINNER + c];
  }
  float sv = siluf_(v);
  xc[idx]  = sv;
  xcb[idx] = f2bf(sv);
}

// ---------- 3-pass chunked linear-recurrence scan ----------
__global__ __launch_bounds__(256) void scan_chunks(const float* __restrict__ ga,
                                                   const float* __restrict__ gb,
                                                   float* __restrict__ Ac,
                                                   float* __restrict__ Bc) {
  int idx = blockIdx.x * 256 + threadIdx.x;  // B*NCHUNK*C = 65536
  int c  = idx & (DINNER - 1);
  int ch = (idx >> 11) & (NCHUNK - 1);
  int bi = idx >> 15;
  size_t base = ((size_t)bi * LSEQ + ch * CLEN) * DINNER + c;
  float A = 1.f, Bv = 0.f;
  for (int t = 0; t < CLEN; ++t) {
    size_t o = base + (size_t)t * DINNER;
    float av = ga[o], bv = gb[o];
    A  = A * av;
    Bv = av * Bv + bv;
  }
  Ac[idx] = A;
  Bc[idx] = Bv;
}

__global__ __launch_bounds__(256) void scan_carry(const float* __restrict__ Ac,
                                                  const float* __restrict__ Bc,
                                                  float* __restrict__ hin) {
  int idx = blockIdx.x * 256 + threadIdx.x;  // B*C = 4096
  int c  = idx & (DINNER - 1);
  int bi = idx >> 11;
  float h = 0.f;
  for (int ch = 0; ch < NCHUNK; ++ch) {
    int o = (bi * NCHUNK + ch) * DINNER + c;
    hin[o] = h;                 // exclusive prefix (state entering chunk)
    h = Ac[o] * h + Bc[o];
  }
}

__global__ __launch_bounds__(256) void scan_apply(const float* __restrict__ ga,
                                                  const float* __restrict__ gb,
                                                  const float* __restrict__ hin,
                                                  const float* __restrict__ z,
                                                  unsigned short* __restrict__ ybf) {
  int idx = blockIdx.x * 256 + threadIdx.x;  // B*NCHUNK*C
  int c  = idx & (DINNER - 1);
  int ch = (idx >> 11) & (NCHUNK - 1);
  int bi = idx >> 15;
  size_t base = ((size_t)bi * LSEQ + ch * CLEN) * DINNER + c;
  float h = hin[idx];
  for (int t = 0; t < CLEN; ++t) {
    size_t o = base + (size_t)t * DINNER;
    h = ga[o] * h + gb[o];
    float zv = z[o];
    ybf[o] = f2bf(h * siluf_(zv));
  }
}

// ---------- launch ----------
extern "C" void kernel_launch(void* const* d_in, const int* in_sizes, int n_in,
                              void* d_out, int out_size, void* d_ws, size_t ws_size,
                              hipStream_t stream) {
  const float* x         = (const float*)d_in[0];
  const float* norm_w    = (const float*)d_in[1];
  const float* norm_b    = (const float*)d_in[2];
  const float* in_proj_w = (const float*)d_in[3];
  const float* conv_w    = (const float*)d_in[4];
  const float* conv_b    = (const float*)d_in[5];
  const float* gate_w    = (const float*)d_in[6];
  const float* gate_b    = (const float*)d_in[7];
  const float* out_proj_w= (const float*)d_in[8];
  float* out = (float*)d_out;

  char* ws = (char*)d_ws;
  size_t off = 0;
  auto take = [&](size_t bytes) -> char* {
    char* p = ws + off;
    off = (off + bytes + 255) & ~(size_t)255;
    return p;
  };
  unsigned short* xn     = (unsigned short*)take((size_t)MTOT * DMODEL * 2);
  unsigned short* w1     = (unsigned short*)take((size_t)2 * DINNER * DMODEL * 2);
  unsigned short* w2     = (unsigned short*)take((size_t)DINNER * DINNER * 2);
  unsigned short* w3     = (unsigned short*)take((size_t)DMODEL * DINNER * 2);
  float* x_inner         = (float*)take((size_t)MTOT * DINNER * 4);
  float* zbuf            = (float*)take((size_t)MTOT * DINNER * 4);
  float* xconv           = (float*)take((size_t)MTOT * DINNER * 4);
  unsigned short* xconvb = (unsigned short*)take((size_t)MTOT * DINNER * 2);
  float* abuf            = (float*)take((size_t)MTOT * DINNER * 4);
  float* bbuf            = (float*)take((size_t)MTOT * DINNER * 4);
  float* Ac              = (float*)take((size_t)BATCH * NCHUNK * DINNER * 4);
  float* Bc              = (float*)take((size_t)BATCH * NCHUNK * DINNER * 4);
  float* hin             = (float*)take((size_t)BATCH * NCHUNK * DINNER * 4);
  unsigned short* ybf    = (unsigned short*)take((size_t)MTOT * DINNER * 2);

  // weights -> bf16
  cvt_bf16<<<(2 * DINNER * DMODEL / 4 + 255) / 256, 256, 0, stream>>>(in_proj_w, w1, 2 * DINNER * DMODEL / 4);
  cvt_bf16<<<(DINNER * DINNER / 4 + 255) / 256, 256, 0, stream>>>(gate_w, w2, DINNER * DINNER / 4);
  cvt_bf16<<<(DMODEL * DINNER / 4 + 255) / 256, 256, 0, stream>>>(out_proj_w, w3, DMODEL * DINNER / 4);

  // layernorm -> bf16
  ln_bf16<<<MTOT / 8, 256, 0, stream>>>(x, norm_w, norm_b, xn, MTOT);

  // in_proj GEMM: [8192,1024] x [4096,1024]^T, split into x_inner / z
  gemm_bf16<0><<<dim3((2 * DINNER) / TN, MTOT / TM), 256, 0, stream>>>(
      xn, w1, MTOT, 2 * DINNER, DMODEL, x_inner, zbuf, nullptr, nullptr);

  // depthwise causal conv + SiLU
  conv_silu<<<(MTOT * DINNER) / 256, 256, 0, stream>>>(x_inner, conv_w, conv_b, xconv, xconvb);

  // gate GEMM: a = sigmoid(xconv @ gate_w^T + gate_b), b = (1-a)*xconv
  gemm_bf16<1><<<dim3(DINNER / TN, MTOT / TM), 256, 0, stream>>>(
      xconvb, w2, MTOT, DINNER, DINNER, abuf, bbuf, gate_b, xconv);

  // chunked scan: h_t = a_t h_{t-1} + b_t ; y = h * silu(z) -> bf16
  scan_chunks<<<(BATCH * NCHUNK * DINNER) / 256, 256, 0, stream>>>(abuf, bbuf, Ac, Bc);
  scan_carry<<<(BATCH * DINNER) / 256, 256, 0, stream>>>(Ac, Bc, hin);
  scan_apply<<<(BATCH * NCHUNK * DINNER) / 256, 256, 0, stream>>>(abuf, bbuf, hin, zbuf, ybf);

  // out_proj GEMM + residual
  gemm_bf16<2><<<dim3(DMODEL / TN, MTOT / TM), 256, 0, stream>>>(
      ybf, w3, MTOT, DMODEL, DINNER, out, nullptr, x, nullptr);
}
